// DilatedResidualBlock_68539088109652
// MI455X (gfx1250) — matrix operand order repaired and checked by reading, not compiled, not run on hardware
//
#include <hip/hip_runtime.h>
#include <hip/hip_bf16.h>
#include <math.h>

typedef __attribute__((ext_vector_type(16))) _Float16 v16h;
typedef __attribute__((ext_vector_type(8)))  float    v8f;
typedef _Float16 h16;

#define NB     2
#define NPTS   20480
#define NK     16
#define NFIN   32
#define ND2    32
#define NF1    64
#define NDOUT  64
#define NPOINT (NB * NPTS)     /* 40960 */
#define EPSV   1e-6f

/* stats slots (floats): [sum C][ssq C][scale C][shift C] */
#define S_BN1 0
#define S_BB1 128
#define S_AP1 256
#define S_BB2 384
#define S_AP2 512
#define S_W2  768
#define S_SC  1280
#define STATS_N 1792

__device__ __forceinline__ float lrelu(float v, float s) { return v < 0.f ? v * s : v; }

__device__ __forceinline__ v8f wmma32(v16h a, v16h b, v8f c) {
  return __builtin_amdgcn_wmma_f32_16x16x32_f16(false, a, false, b, (short)0, c, false, false);
}

/* A operand (16x32 f16): lane<16 row M=lane, half h -> K=h;
   lanes>=16 add +16 to K. row = LDS pointer to row M, f0 = logical K offset.
   The 16 halfs are contiguous -> compiler emits 2x ds_load_b128. */
__device__ __forceinline__ v16h packA(const h16* row, int f0, int hi) {
  v16h a;
#pragma unroll
  for (int h = 0; h < 16; ++h)
    a[h] = row[f0 + h + (hi << 4)];
  return a;
}
/* B operand from pre-swizzled LDS table: lane's 16 halfs contiguous, 32B aligned. */
__device__ __forceinline__ v16h loadB(const h16* sw, int slot, int lane) {
  return *(const v16h*)(sw + ((slot << 5) + lane) * 16);
}

/* ---------------- small utility kernels ---------------- */

__global__ void kzero(float* g) {
  int i = blockIdx.x * blockDim.x + threadIdx.x;
  if (i < STATS_N) g[i] = 0.f;
}

__global__ void kfin(float* g, int base, int C, float invN, const float* gamma, const float* beta) {
  int c = threadIdx.x;
  if (c >= C) return;
  float m = g[base + c] * invN;
  float v = fmaxf(g[base + C + c] * invN - m * m, 0.f);
  float sc = gamma[c] * rsqrtf(v + EPSV);
  g[base + 2 * C + c] = sc;
  g[base + 3 * C + c] = beta[c] - m * sc;
}

/* pre-activation of f_in @ W(32xCout) + bias for one (b,n,c) */
__device__ __forceinline__ float pre_fc32(const float* feat, const float* W, const float* bias,
                                          int b, int n, int c, int Cout) {
  float p = bias[c];
#pragma unroll 8
  for (int f = 0; f < NFIN; ++f)
    p += feat[((size_t)(b * NFIN + f)) * NPTS + n] * W[f * Cout + c];
  return p;
}

__global__ void kstats_bn1(const float* feat, const float* w1, const float* b1, float* g) {
  __shared__ float ls[64];
  if (threadIdx.x < 64) ls[threadIdx.x] = 0.f;
  __syncthreads();
  int gid = blockIdx.x * blockDim.x + threadIdx.x;
  int c = gid & 31, p = gid >> 5;
  int b = p / NPTS, n = p % NPTS;
  float v = pre_fc32(feat, w1, b1, b, n, c, ND2);
  atomicAdd(&ls[c], v);
  atomicAdd(&ls[32 + c], v * v);
  __syncthreads();
  if (threadIdx.x < 32) {
    atomicAdd(&g[S_BN1 + threadIdx.x], ls[threadIdx.x]);
    atomicAdd(&g[S_BN1 + 32 + threadIdx.x], ls[32 + threadIdx.x]);
  }
}

__global__ void kapply_fpc(const float* feat, const float* w1, const float* b1,
                           const float* g, h16* fpc) {
  int gid = blockIdx.x * blockDim.x + threadIdx.x;
  int c = gid & 31, p = gid >> 5;
  int b = p / NPTS, n = p % NPTS;
  float v = pre_fc32(feat, w1, b1, b, n, c, ND2);
  v = lrelu(v * g[S_BN1 + 64 + c] + g[S_BN1 + 96 + c], 0.2f);
  fpc[(size_t)p * ND2 + c] = (h16)v;
}

/* geometry pre-activation (10 -> 32) for one (b,n,k,c) */
__device__ __forceinline__ float geo_pre(const float* xyz, int bPN, int n, int idx,
                                         const float* bw, const float* bb, int c) {
  const float* cp = xyz + (size_t)(bPN + n) * 3;
  const float* np = xyz + (size_t)(bPN + idx) * 3;
  float cx = cp[0], cy = cp[1], cz = cp[2];
  float nx = np[0], ny = np[1], nz = np[2];
  float rx = cx - nx, ry = cy - ny, rz = cz - nz;
  float d = sqrtf(rx * rx + ry * ry + rz * rz);
  float gg[10] = {d, rx, ry, rz, cx, cy, cz, nx, ny, nz};
  float p = bb[c];
#pragma unroll
  for (int j = 0; j < 10; ++j) p += gg[j] * bw[j * ND2 + c];
  return p;
}

__global__ void kstats_bb1(const float* xyz, const int* neigh,
                           const float* bw, const float* bb, float* g) {
  __shared__ float ls[64];
  if (threadIdx.x < 64) ls[threadIdx.x] = 0.f;
  __syncthreads();
  int gid = blockIdx.x * blockDim.x + threadIdx.x;
  int c = gid & 31, k = (gid >> 5) & 15, p = gid >> 9;
  int b = p / NPTS, n = p % NPTS;
  int idx = neigh[(size_t)p * NK + k];
  float v = geo_pre(xyz, b * NPTS, n, idx, bw, bb, c);
  atomicAdd(&ls[c], v);
  atomicAdd(&ls[32 + c], v * v);
  __syncthreads();
  if (threadIdx.x < 32) {
    atomicAdd(&g[S_BB1 + threadIdx.x], ls[threadIdx.x]);
    atomicAdd(&g[S_BB1 + 32 + threadIdx.x], ls[32 + threadIdx.x]);
  }
}

/* stage bn+lrelu geometry features into LDS tile cols 32..63 (c = lane 0..31) */
__device__ __forceinline__ void stage_fxyz(h16* X, int ldx, const float* xyz, const int* nb,
                                           int bPN, int n, const float* bw1, const float* bb1,
                                           const float* sc, const float* sh, int c) {
  for (int k = 0; k < NK; ++k) {
    int idx = nb[k];
    float p = geo_pre(xyz, bPN, n, idx, bw1, bb1, c);
    p = lrelu(p * sc[c] + sh[c], 0.2f);
    X[k * ldx + 32 + c] = (h16)p;
  }
}

/* WMMA attention: logits(16x64) = X(16x64) @ fc^T, softmax over neighbors, agg[64] out.
   sFC is the pre-swizzled B-operand table: slot = tile*2 + kstep. */
__device__ __forceinline__ void attn_agg(const h16* xrow, const h16* X, int ldx,
                                         const h16* sFC, float* aggOut, int lane) {
  const int hi = lane >> 4, l15 = lane & 15;
  v16h a0 = packA(xrow, 0, hi);
  v16h a1 = packA(xrow, 32, hi);
#pragma unroll
  for (int t = 0; t < 4; ++t) {
    int col = 16 * t + l15;
    v8f c8 = {0.f, 0.f, 0.f, 0.f, 0.f, 0.f, 0.f, 0.f};
    c8 = wmma32(a0, loadB(sFC, 2 * t + 0, lane), c8);
    c8 = wmma32(a1, loadB(sFC, 2 * t + 1, lane), c8);
    float m = c8[0];
#pragma unroll
    for (int i = 1; i < 8; ++i) m = fmaxf(m, c8[i]);
    m = fmaxf(m, __shfl_xor(m, 16, 32));
    float e[8], s = 0.f;
#pragma unroll
    for (int i = 0; i < 8; ++i) { e[i] = __expf(c8[i] - m); s += e[i]; }
    s += __shfl_xor(s, 16, 32);
    float inv = 1.f / s;
    float p = 0.f;
#pragma unroll
    for (int i = 0; i < 8; ++i)
      p += (float)X[(i + 8 * hi) * ldx + col] * e[i] * inv;
    p += __shfl_xor(p, 16, 32);
    if (hi == 0) aggOut[col] = p;
  }
}

/* fill swizzled B table for the 64x64 fc (logits[k,o] = sum_f x[k,f]*fc[o,f]):
   slot = tile*2+step; element (slot, lane, h) = fc[o*64 + K],
   K = step*32 + h + (lane>=16)*16, o = tile*16 + (lane&15). */
__device__ __forceinline__ void fill_swFC(h16* sFC, const float* fc, int tid) {
  for (int i = tid; i < NF1 * NF1; i += 256) {
    int h = i & 15, ln = (i >> 4) & 31, ts = i >> 9;
    int K = (ts & 1) * 32 + h + (ln >> 4) * 16;
    int o = (ts >> 1) * 16 + (ln & 15);
    sFC[i] = (h16)fc[o * NF1 + K];
  }
}
/* fill swizzled B table for bb_w2 (32x32, B[K=f][N=c] = bw2[f*32+c]); slot = tile. */
__device__ __forceinline__ void fill_swB2(h16* sB2, const float* bw2, int tid) {
  for (int i = tid; i < ND2 * ND2; i += 256) {
    int h = i & 15, ln = (i >> 4) & 31, tile = i >> 9;
    int K = h + (ln >> 4) * 16;
    int c = tile * 16 + (ln & 15);
    sB2[i] = (h16)bw2[K * ND2 + c];
  }
}

/* ---------------- attention-pool kernel 1 ---------------- */
__global__ __launch_bounds__(256) void katt1(
    const h16* __restrict__ fpc, const float* __restrict__ xyz, const int* __restrict__ neigh,
    const float* __restrict__ fc, const float* __restrict__ bw1, const float* __restrict__ bb1,
    const float* __restrict__ bw2, const float* __restrict__ bb2,
    const float* __restrict__ apw, const float* __restrict__ apb,
    float* __restrict__ g, float* __restrict__ z1) {
  __shared__ h16 sFC[NF1 * NF1];
  __shared__ h16 sB2[ND2 * ND2];
  __shared__ h16 sX[8][NK][72];
  __shared__ float sAgg[8][NF1];
  __shared__ float sStat[256];  /* 0:64 apsum 64:128 apssq 128:160 b2sum 160:192 b2ssq */

  fill_swFC(sFC, fc, threadIdx.x);
  fill_swB2(sB2, bw2, threadIdx.x);
  sStat[threadIdx.x] = 0.f;
  __syncthreads();

  const int lane = threadIdx.x & 31, wv = threadIdx.x >> 5;
  const int hi = lane >> 4, l15 = lane & 15;
  const int point = blockIdx.x * 8 + wv;
  const int b = point / NPTS, n = point % NPTS, bPN = b * NPTS;
  const int* nb = neigh + (size_t)point * NK;
  h16* X = &sX[wv][0][0];
  const int ldx = 72;

  for (int t = lane; t < NK * ND2; t += 32) {
    int k = t >> 5, c = t & 31;
    X[k * ldx + c] = fpc[((size_t)bPN + nb[k]) * ND2 + c];
  }
  stage_fxyz(X, ldx, xyz, nb, bPN, n, bw1, bb1, g + S_BB1 + 64, g + S_BB1 + 96, lane);
  __syncthreads();

  const h16* xrow = X + l15 * ldx;
  attn_agg(xrow, X, ldx, sFC, sAgg[wv], lane);
  __syncthreads();

  { /* z = agg @ ap1_w + b, plus ap1 stats */
    int c = lane;
    float zz = apb[c];
    for (int f = 0; f < NF1; ++f) zz += sAgg[wv][f] * apw[f * ND2 + c];
    z1[(size_t)point * ND2 + c] = zz;
    atomicAdd(&sStat[c], zz);
    atomicAdd(&sStat[64 + c], zz * zz);
  }
  { /* bb2 pre-activation stats via WMMA: A = f_xyz cols 32..63 */
    v16h a = packA(xrow, 32, hi);
#pragma unroll
    for (int t = 0; t < 2; ++t) {
      int ch = 16 * t + l15;
      v8f c8 = {0.f, 0.f, 0.f, 0.f, 0.f, 0.f, 0.f, 0.f};
      c8 = wmma32(a, loadB(sB2, t, lane), c8);
      float bias = bb2[ch];
      float s = 0.f, q = 0.f;
#pragma unroll
      for (int i = 0; i < 8; ++i) { float v = c8[i] + bias; s += v; q += v * v; }
      s += __shfl_xor(s, 16, 32);
      q += __shfl_xor(q, 16, 32);
      if (hi == 0) { atomicAdd(&sStat[128 + ch], s); atomicAdd(&sStat[160 + ch], q); }
    }
  }
  __syncthreads();
  if (threadIdx.x < 32) {
    int c = threadIdx.x;
    atomicAdd(&g[S_AP1 + c], sStat[c]);
    atomicAdd(&g[S_AP1 + 32 + c], sStat[64 + c]);
    atomicAdd(&g[S_BB2 + c], sStat[128 + c]);
    atomicAdd(&g[S_BB2 + 32 + c], sStat[160 + c]);
  }
}

__global__ void kapply_agg1(const float* z1, const float* g, h16* ag1) {
  int gid = blockIdx.x * blockDim.x + threadIdx.x;
  int c = gid & 31, p = gid >> 5;
  float v = z1[(size_t)p * ND2 + c];
  v = lrelu(v * g[S_AP1 + 64 + c] + g[S_AP1 + 96 + c], 0.2f);
  ag1[(size_t)p * ND2 + c] = (h16)v;
}

/* ---------------- attention-pool kernel 2 ---------------- */
__global__ __launch_bounds__(256) void katt2(
    const h16* __restrict__ ag1, const float* __restrict__ xyz, const int* __restrict__ neigh,
    const float* __restrict__ fc2, const float* __restrict__ bw1, const float* __restrict__ bb1,
    const float* __restrict__ bw2, const float* __restrict__ bb2,
    const float* __restrict__ apw2, const float* __restrict__ apb2,
    float* __restrict__ g, float* __restrict__ z2) {
  __shared__ h16 sFC[NF1 * NF1];
  __shared__ h16 sB2[ND2 * ND2];
  __shared__ h16 sX[8][NK][72];
  __shared__ float sAgg[8][NF1];
  __shared__ float sStat[256];

  fill_swFC(sFC, fc2, threadIdx.x);
  fill_swB2(sB2, bw2, threadIdx.x);
  sStat[threadIdx.x] = 0.f;
  __syncthreads();

  const int lane = threadIdx.x & 31, wv = threadIdx.x >> 5;
  const int hi = lane >> 4, l15 = lane & 15;
  const int point = blockIdx.x * 8 + wv;
  const int b = point / NPTS, n = point % NPTS, bPN = b * NPTS;
  const int* nb = neigh + (size_t)point * NK;
  h16* X = &sX[wv][0][0];
  const int ldx = 72;
  const h16* xrow = X + l15 * ldx;

  /* f_xyz (bn_bb1 + lrelu) into cols 32..63 */
  stage_fxyz(X, ldx, xyz, nb, bPN, n, bw1, bb1, g + S_BB1 + 64, g + S_BB1 + 96, lane);

  { /* f_xyz2 = lrelu(bn_bb2(f_xyz @ bb_w2)) written back into cols 32..63 (WMMA) */
    v16h a = packA(xrow, 32, hi);
    v8f t0 = {0.f, 0.f, 0.f, 0.f, 0.f, 0.f, 0.f, 0.f};
    v8f t1 = {0.f, 0.f, 0.f, 0.f, 0.f, 0.f, 0.f, 0.f};
    t0 = wmma32(a, loadB(sB2, 0, lane), t0);
    t1 = wmma32(a, loadB(sB2, 1, lane), t1);
    const float* sc = g + S_BB2 + 64;
    const float* sh = g + S_BB2 + 96;
#pragma unroll
    for (int i = 0; i < 8; ++i) {
      int M = i + 8 * hi;
      float v0 = lrelu((t0[i] + bb2[l15]) * sc[l15] + sh[l15], 0.2f);
      float v1 = lrelu((t1[i] + bb2[16 + l15]) * sc[16 + l15] + sh[16 + l15], 0.2f);
      X[M * ldx + 32 + l15] = (h16)v0;
      X[M * ldx + 48 + l15] = (h16)v1;
    }
  }
  /* gather agg1 neighbors into cols 0..31 */
  for (int t = lane; t < NK * ND2; t += 32) {
    int k = t >> 5, c = t & 31;
    X[k * ldx + c] = ag1[((size_t)bPN + nb[k]) * ND2 + c];
  }
  __syncthreads();

  attn_agg(xrow, X, ldx, sFC, sAgg[wv], lane);
  __syncthreads();

#pragma unroll
  for (int r = 0; r < 2; ++r) { /* z2 = agg @ ap2_w + b (64 ch) + stats */
    int c = lane + 32 * r;
    float zz = apb2[c];
    for (int f = 0; f < NF1; ++f) zz += sAgg[wv][f] * apw2[f * NDOUT + c];
    z2[(size_t)point * NDOUT + c] = zz;
    atomicAdd(&sStat[c], zz);
    atomicAdd(&sStat[64 + c], zz * zz);
  }
  __syncthreads();
  if (threadIdx.x < 64) {
    atomicAdd(&g[S_AP2 + threadIdx.x], sStat[threadIdx.x]);
    atomicAdd(&g[S_AP2 + 64 + threadIdx.x], sStat[64 + threadIdx.x]);
  }
}

/* main/shortcut pre-activations + stats (grid-stride, register-local stat accumulation) */
__global__ __launch_bounds__(128) void kfinal_pre(
    const float* __restrict__ feat, const float* __restrict__ z2, float* __restrict__ g,
    const float* __restrict__ w2, const float* __restrict__ b2,
    const float* __restrict__ scw, const float* __restrict__ scb,
    float* __restrict__ mp, float* __restrict__ sp) {
  __shared__ float a2s[NDOUT];
  int c = threadIdx.x;
  float sm = 0.f, qm = 0.f, ss = 0.f, qs = 0.f;
  for (int p = blockIdx.x; p < NPOINT; p += gridDim.x) {
    int b = p / NPTS, n = p % NPTS;
    __syncthreads();
    if (c < NDOUT) {
      float v = z2[(size_t)p * NDOUT + c];
      a2s[c] = lrelu(v * g[S_AP2 + 128 + c] + g[S_AP2 + 192 + c], 0.2f);
    }
    __syncthreads();
    float m = b2[c];
    for (int f = 0; f < NDOUT; ++f) m += a2s[f] * w2[f * 128 + c];
    float s = scb[c];
    for (int f = 0; f < NFIN; ++f) s += feat[((size_t)(b * NFIN + f)) * NPTS + n] * scw[f * 128 + c];
    mp[(size_t)p * 128 + c] = m;
    sp[(size_t)p * 128 + c] = s;
    sm += m; qm += m * m; ss += s; qs += s * s;
  }
  atomicAdd(&g[S_W2 + c], sm);
  atomicAdd(&g[S_W2 + 128 + c], qm);
  atomicAdd(&g[S_SC + c], ss);
  atomicAdd(&g[S_SC + 128 + c], qs);
}

__global__ void kout(const float* mp, const float* sp, const float* g, float* out) {
  int gid = blockIdx.x * blockDim.x + threadIdx.x;
  int c = gid & 127, p = gid >> 7;
  int b = p / NPTS, n = p % NPTS;
  float v = mp[(size_t)p * 128 + c] * g[S_W2 + 256 + c] + g[S_W2 + 384 + c]
          + sp[(size_t)p * 128 + c] * g[S_SC + 256 + c] + g[S_SC + 384 + c];
  out[((size_t)(b * 128 + c)) * NPTS + n] = lrelu(v, 0.01f);
}

extern "C" void kernel_launch(void* const* d_in, const int* in_sizes, int n_in,
                              void* d_out, int out_size, void* d_ws, size_t ws_size,
                              hipStream_t stream) {
  const float* feat  = (const float*)d_in[0];
  const float* xyz   = (const float*)d_in[1];
  const int*   nidx  = (const int*)d_in[2];
  const float* w1    = (const float*)d_in[3];
  const float* b1    = (const float*)d_in[4];
  const float* g1    = (const float*)d_in[5];
  const float* be1   = (const float*)d_in[6];
  const float* bbw1  = (const float*)d_in[7];
  const float* bbb1  = (const float*)d_in[8];
  const float* bbg1  = (const float*)d_in[9];
  const float* bbbe1 = (const float*)d_in[10];
  const float* ap1fc = (const float*)d_in[11];
  const float* ap1w  = (const float*)d_in[12];
  const float* ap1b  = (const float*)d_in[13];
  const float* ap1g  = (const float*)d_in[14];
  const float* ap1be = (const float*)d_in[15];
  const float* bbw2  = (const float*)d_in[16];
  const float* bbb2  = (const float*)d_in[17];
  const float* bbg2  = (const float*)d_in[18];
  const float* bbbe2 = (const float*)d_in[19];
  const float* ap2fc = (const float*)d_in[20];
  const float* ap2w  = (const float*)d_in[21];
  const float* ap2b  = (const float*)d_in[22];
  const float* ap2g  = (const float*)d_in[23];
  const float* ap2be = (const float*)d_in[24];
  const float* w2    = (const float*)d_in[25];
  const float* b2    = (const float*)d_in[26];
  const float* g2    = (const float*)d_in[27];
  const float* be2   = (const float*)d_in[28];
  const float* scw   = (const float*)d_in[29];
  const float* scb   = (const float*)d_in[30];
  const float* scg   = (const float*)d_in[31];
  const float* scbe  = (const float*)d_in[32];

  char* ws = (char*)d_ws;
  float* g = (float*)ws;
  size_t off = 2048 * sizeof(float);
  float* z1 = (float*)(ws + off); off += (size_t)NPOINT * ND2 * 4;
  float* z2 = (float*)(ws + off); off += (size_t)NPOINT * NDOUT * 4;
  float* mp = (float*)(ws + off); off += (size_t)NPOINT * 128 * 4;
  float* sp = (float*)(ws + off); off += (size_t)NPOINT * 128 * 4;
  h16* fpc  = (h16*)(ws + off);   off += (size_t)NPOINT * ND2 * 2;
  h16* ag1  = (h16*)(ws + off);
  float* out = (float*)d_out;

  kzero<<<8, 256, 0, stream>>>(g);
  kstats_bn1<<<NPOINT * ND2 / 256, 256, 0, stream>>>(feat, w1, b1, g);
  kstats_bb1<<<NPOINT * NK * ND2 / 256, 256, 0, stream>>>(xyz, nidx, bbw1, bbb1, g);
  kfin<<<1, 128, 0, stream>>>(g, S_BN1, 32, 1.f / NPOINT, g1, be1);
  kfin<<<1, 128, 0, stream>>>(g, S_BB1, 32, 1.f / ((float)NPOINT * NK), bbg1, bbbe1);
  kapply_fpc<<<NPOINT * ND2 / 256, 256, 0, stream>>>(feat, w1, b1, g, fpc);
  katt1<<<NPOINT / 8, 256, 0, stream>>>(fpc, xyz, nidx, ap1fc, bbw1, bbb1, bbw2, bbb2,
                                        ap1w, ap1b, g, z1);
  kfin<<<1, 128, 0, stream>>>(g, S_AP1, 32, 1.f / NPOINT, ap1g, ap1be);
  kfin<<<1, 128, 0, stream>>>(g, S_BB2, 32, 1.f / ((float)NPOINT * NK), bbg2, bbbe2);
  kapply_agg1<<<NPOINT * ND2 / 256, 256, 0, stream>>>(z1, g, ag1);
  katt2<<<NPOINT / 8, 256, 0, stream>>>(ag1, xyz, nidx, ap2fc, bbw1, bbb1, bbw2, bbb2,
                                        ap2w, ap2b, g, z2);
  kfin<<<1, 128, 0, stream>>>(g, S_AP2, 64, 1.f / NPOINT, ap2g, ap2be);
  kfinal_pre<<<512, 128, 0, stream>>>(feat, z2, g, w2, b2, scw, scb, mp, sp);
  kfin<<<1, 128, 0, stream>>>(g, S_W2, 128, 1.f / NPOINT, g2, be2);
  kfin<<<1, 128, 0, stream>>>(g, S_SC, 128, 1.f / NPOINT, scg, scbe);
  kout<<<NPOINT * 128 / 256, 256, 0, stream>>>(mp, sp, g, out);
}
